// TFTBlock_2302102471033
// MI455X (gfx1250) — compile-verified, matrix operand order checked by reading, not run on hardware
//
#include <hip/hip_runtime.h>
#include <hip/hip_bf16.h>
#include <math.h>

// ---------------- problem constants ----------------
#define Bb   2
#define Tt   2048
#define Cc   1024
#define Hh   16
#define Dd   64
#define DFFf 4096
#define BT   (Bb*Tt)          // 4096 rows

typedef __bf16 bf16_t;
typedef bf16_t bf16x16 __attribute__((ext_vector_type(16)));
typedef bf16_t bf16x8  __attribute__((ext_vector_type(8)));
typedef float  f32x8   __attribute__((ext_vector_type(8)));

#define SHUF16(lo, hi) __builtin_shufflevector((lo), (hi), 0,1,2,3,4,5,6,7,8,9,10,11,12,13,14,15)
#define NEG_BIG (-3.0e38f)

// ----- CDNA5 async global->LDS copy (ASYNCcnt-tracked, no VGPR round-trip) --
__device__ __forceinline__ void async_b128(const void* lds, const void* g) {
    asm volatile("global_load_async_to_lds_b128 %0, %1, off"
                 :: "v"((uint32_t)(uintptr_t)lds),
                    "v"((uint64_t)(uintptr_t)g)
                 : "memory");
}
#define WAIT_ASYNC() asm volatile("s_wait_asynccnt 0" ::: "memory")
#define WAIT_DS0()   asm volatile("s_wait_dscnt 0"    ::: "memory")

// ---------------- f32 -> bf16 convert ----------------
__global__ void k_cvt_bf16(const float* __restrict__ src, bf16_t* __restrict__ dst, int n) {
    int i = blockIdx.x * blockDim.x + threadIdx.x;
    if (i < n) dst[i] = (bf16_t)src[i];
}

// ---------------- fused add + layernorm -> bf16 ----------------
__global__ __launch_bounds__(256)
void k_add_ln(const float* __restrict__ xa, const float* __restrict__ xb,
              const float* __restrict__ w, const float* __restrict__ bia,
              bf16_t* __restrict__ out) {
    __shared__ float s1[256], s2[256];
    int row = blockIdx.x;
    int tid = threadIdx.x;
    const float* a = xa + (size_t)row * Cc;
    const float* b = xb + (size_t)row * Cc;
    float x[4]; float s = 0.f, q = 0.f;
#pragma unroll
    for (int i = 0; i < 4; i++) {
        int c = tid + i * 256;
        x[i] = a[c] + b[c];
        s += x[i]; q += x[i] * x[i];
    }
    s1[tid] = s; s2[tid] = q;
    __syncthreads();
    for (int m = 128; m > 0; m >>= 1) {
        if (tid < m) { s1[tid] += s1[tid + m]; s2[tid] += s2[tid + m]; }
        __syncthreads();
    }
    float mu = s1[0] * (1.0f / Cc);
    float var = s2[0] * (1.0f / Cc) - mu * mu;
    float rs = rsqrtf(var + 1e-5f);
    bf16_t* o = out + (size_t)row * Cc;
#pragma unroll
    for (int i = 0; i < 4; i++) {
        int c = tid + i * 256;
        o[c] = (bf16_t)((x[i] - mu) * rs * w[c] + bia[c]);
    }
}

// ---------------- WMMA GEMM: D[n,m] = sum_k A[n,k]*Bw[m,k] + bias ----------
// 128x64 tile, BK=32, double-buffered LDS filled by async copies.
// mode 0: scatter q (scaled) / k to [B,H,T,D] bf16
// mode 1: GELU -> bf16 row-major
// mode 2: + res -> f32 row-major
__global__ __launch_bounds__(256)
void k_gemm(const bf16_t* __restrict__ A, const bf16_t* __restrict__ Bw,
            const float* __restrict__ bias, int N, int M, int K,
            int mode, float qscale,
            bf16_t* __restrict__ qout, bf16_t* __restrict__ kout,
            bf16_t* __restrict__ bout,
            const float* __restrict__ res, float* __restrict__ fout) {
    __shared__ __align__(16) bf16_t As[2][128][32];   // 16 KB
    __shared__ __align__(16) bf16_t Bs[2][64][32];    //  8 KB

    int tid = threadIdx.x;
    int wid = tid >> 5, lane = tid & 31;
    int half = lane >> 4, l16 = lane & 15;
    int nbase = blockIdx.x * 128;
    int mbase = blockIdx.y * 64;

    // staging geometry: each thread moves two 8-elem A chunks + one B chunk
    int sr = tid >> 2;            // 0..63
    int sc = (tid & 3) * 8;       // 0,8,16,24
    const bf16_t* Ag0 = A  + (size_t)(nbase + sr)      * K + sc;
    const bf16_t* Ag1 = A  + (size_t)(nbase + sr + 64) * K + sc;
    const bf16_t* Bg  = Bw + (size_t)(mbase + sr)      * K + sc;

    f32x8 acc0 = {}, acc1 = {}, acc2 = {}, acc3 = {};
    int nk = K >> 5;

    // prologue: fill buffer 0
    async_b128(&As[0][sr][sc],      Ag0);
    async_b128(&As[0][sr + 64][sc], Ag1);
    async_b128(&Bs[0][sr][sc],      Bg);

    for (int i = 0; i < nk; i++) {
        int buf = i & 1;
        WAIT_ASYNC();
        __syncthreads();
        if (i + 1 < nk) {
            int k0 = (i + 1) << 5;
            async_b128(&As[buf ^ 1][sr][sc],      Ag0 + k0);
            async_b128(&As[buf ^ 1][sr + 64][sc], Ag1 + k0);
            async_b128(&Bs[buf ^ 1][sr][sc],      Bg  + k0);
        }
        // A fragment (16x32): rows wid*16 + l16
        bf16x8 lo = *(const bf16x8*)&As[buf][wid * 16 + l16][half * 8];
        bf16x8 hi = *(const bf16x8*)&As[buf][wid * 16 + l16][16 + half * 8];
        bf16x16 af = SHUF16(lo, hi);
        // 4 B fragments (32x16 each) covering 64 output cols
#pragma unroll
        for (int nb = 0; nb < 4; nb++) {
            bf16x8 blo = *(const bf16x8*)&Bs[buf][nb * 16 + l16][half * 8];
            bf16x8 bhi = *(const bf16x8*)&Bs[buf][nb * 16 + l16][16 + half * 8];
            bf16x16 bfv = SHUF16(blo, bhi);
            if (nb == 0) acc0 = __builtin_amdgcn_wmma_f32_16x16x32_bf16(false, af, false, bfv, (short)0, acc0, false, false);
            if (nb == 1) acc1 = __builtin_amdgcn_wmma_f32_16x16x32_bf16(false, af, false, bfv, (short)0, acc1, false, false);
            if (nb == 2) acc2 = __builtin_amdgcn_wmma_f32_16x16x32_bf16(false, af, false, bfv, (short)0, acc2, false, false);
            if (nb == 3) acc3 = __builtin_amdgcn_wmma_f32_16x16x32_bf16(false, af, false, bfv, (short)0, acc3, false, false);
        }
        __syncthreads();
    }

#pragma unroll
    for (int r = 0; r < 8; r++) {
        int row = nbase + wid * 16 + 8 * half + r;
        float vv[4] = { acc0[r], acc1[r], acc2[r], acc3[r] };
#pragma unroll
        for (int u = 0; u < 4; u++) {
            int c = mbase + u * 16 + l16;
            float v = vv[u] + bias[c];
            if (mode == 0) {
                int b_ = row / Tt, t = row - b_ * Tt;
                int cd = (c < Cc) ? c : c - Cc;
                int hh = cd / Dd, d = cd - hh * Dd;
                size_t o = (((size_t)b_ * Hh + hh) * Tt + t) * Dd + d;
                if (c < Cc) qout[o] = (bf16_t)(v * qscale);
                else        kout[o] = (bf16_t)v;
            } else if (mode == 1) {
                float g = 0.5f * v * (1.0f + erff(v * 0.70710678118f));
                bout[(size_t)row * M + c] = (bf16_t)g;
            } else {
                size_t o = (size_t)row * M + c;
                fout[o] = v + res[o];
            }
        }
    }
}

// ---------------- v = head-mix of xt: v[b,h,t,d] = sum_j vfact[h,j]*xt[b,t,j,d]
__global__ void k_vproj(const float* __restrict__ xt, const float* __restrict__ vf,
                        bf16_t* __restrict__ vout) {
    size_t i = (size_t)blockIdx.x * blockDim.x + threadIdx.x;   // over B*T*H*D
    int d  = (int)(i & (Dd - 1));
    int hh = (int)((i >> 6) & (Hh - 1));
    int bt = (int)(i >> 10);                                    // D*H = 1024
    const float* xrow = xt + (size_t)bt * Cc;
    float acc = 0.f;
#pragma unroll
    for (int j = 0; j < Hh; j++) acc += vf[hh * Hh + j] * xrow[j * Dd + d];
    int b_ = bt / Tt, t = bt - b_ * Tt;
    vout[(((size_t)b_ * Hh + hh) * Tt + t) * Dd + d] = (bf16_t)acc;
}

// ---------------- flash attention per (b,h), ALiBi + causal ----------------
// Q,K,V: bf16 [B,H,T,D] (q pre-scaled by 1/sqrt(D)); Y: f32 [B,T,H,D]
__global__ __launch_bounds__(128)
void k_attn(const bf16_t* __restrict__ Q, const bf16_t* __restrict__ Kt,
            const bf16_t* __restrict__ Vt, float* __restrict__ Y) {
    __shared__ __align__(16) bf16_t Ks[2][32][64];    // [key][d], 8 KB
    __shared__ __align__(16) bf16_t Vs[2][32][64];    // [key][d], 8 KB
    __shared__ __align__(16) bf16_t Ps[4][16][32];    // per-wave P tiles, 4 KB

    int bh = blockIdx.y;
    int h  = bh & (Hh - 1);
    int b_ = bh >> 4;
    int qblock = blockIdx.x * 64;
    int tid = threadIdx.x, wid = tid >> 5, lane = tid & 31;
    int half = lane >> 4, l16 = lane & 15;
    int qbase = qblock + wid * 16;

    const bf16_t* Qh = Q  + (size_t)bh * Tt * Dd;
    const bf16_t* Kh = Kt + (size_t)bh * Tt * Dd;
    const bf16_t* Vh = Vt + (size_t)bh * Tt * Dd;

    float slope = exp2f(-0.5f * (float)(h + 1));

    // Q fragments for this wave's 16 rows (D=64 => 2 K-chunks of 32)
    bf16x16 qf[2];
    {
        const bf16_t* qrow = Qh + (size_t)(qbase + l16) * Dd;
#pragma unroll
        for (int c = 0; c < 2; c++) {
            bf16x8 lo = *(const bf16x8*)(qrow + c * 32 + half * 8);
            bf16x8 hi = *(const bf16x8*)(qrow + c * 32 + 16 + half * 8);
            qf[c] = SHUF16(lo, hi);
        }
    }

    // staging geometry: 2 async b128 per thread per matrix per tile
    int kr0 = tid >> 3;               // 0..15
    int kr1 = kr0 + 16;               // 16..31
    int dc  = (tid & 7) * 8;          // 0..56

    float mrow[8], lrow[8];
#pragma unroll
    for (int r = 0; r < 8; r++) { mrow[r] = NEG_BIG; lrow[r] = 0.f; }
    f32x8 y0 = {}, y1 = {}, y2 = {}, y3 = {};

    int jend = qblock + 64;
    // prologue: fill buffer 0 with keys [0,32)
    async_b128(&Ks[0][kr0][dc], Kh + (size_t)kr0 * Dd + dc);
    async_b128(&Ks[0][kr1][dc], Kh + (size_t)kr1 * Dd + dc);
    async_b128(&Vs[0][kr0][dc], Vh + (size_t)kr0 * Dd + dc);
    async_b128(&Vs[0][kr1][dc], Vh + (size_t)kr1 * Dd + dc);

    for (int j0 = 0; j0 < jend; j0 += 32) {
        int buf = (j0 >> 5) & 1;
        WAIT_ASYNC();
        __syncthreads();
        if (j0 + 32 < jend) {
            int jn = j0 + 32;
            async_b128(&Ks[buf ^ 1][kr0][dc], Kh + (size_t)(jn + kr0) * Dd + dc);
            async_b128(&Ks[buf ^ 1][kr1][dc], Kh + (size_t)(jn + kr1) * Dd + dc);
            async_b128(&Vs[buf ^ 1][kr0][dc], Vh + (size_t)(jn + kr0) * Dd + dc);
            async_b128(&Vs[buf ^ 1][kr1][dc], Vh + (size_t)(jn + kr1) * Dd + dc);
        }

        if (j0 <= qbase + 15) {   // wave-uniform: tile not fully masked
            // scores: S0 = keys j0..j0+15, S1 = j0+16..j0+31
            f32x8 s0 = {}, s1 = {};
#pragma unroll
            for (int c = 0; c < 2; c++) {
                bf16x8 lo = *(const bf16x8*)&Ks[buf][l16][c * 32 + half * 8];
                bf16x8 hi = *(const bf16x8*)&Ks[buf][l16][c * 32 + 16 + half * 8];
                bf16x16 kf0 = SHUF16(lo, hi);
                lo = *(const bf16x8*)&Ks[buf][16 + l16][c * 32 + half * 8];
                hi = *(const bf16x8*)&Ks[buf][16 + l16][c * 32 + 16 + half * 8];
                bf16x16 kf1 = SHUF16(lo, hi);
                s0 = __builtin_amdgcn_wmma_f32_16x16x32_bf16(false, qf[c], false, kf0, (short)0, s0, false, false);
                s1 = __builtin_amdgcn_wmma_f32_16x16x32_bf16(false, qf[c], false, kf1, (short)0, s1, false, false);
            }
            // online softmax update (row lives across the 16 lanes of this half)
#pragma unroll
            for (int r = 0; r < 8; r++) {
                float qrow = (float)(qbase + 8 * half + r);
                float kc0 = (float)(j0 + l16), kc1 = kc0 + 16.f;
                float a0 = (kc0 <= qrow) ? s0[r] + slope * (kc0 - qrow) : NEG_BIG;
                float a1 = (kc1 <= qrow) ? s1[r] + slope * (kc1 - qrow) : NEG_BIG;
                float tm = fmaxf(a0, a1);
#pragma unroll
                for (int m = 1; m < 16; m <<= 1) tm = fmaxf(tm, __shfl_xor(tm, m, 32));
                float mnew = fmaxf(mrow[r], tm);
                float scl = __expf(mrow[r] - mnew);
                float p0 = __expf(a0 - mnew);
                float p1 = __expf(a1 - mnew);
                float ps = p0 + p1;
#pragma unroll
                for (int m = 1; m < 16; m <<= 1) ps += __shfl_xor(ps, m, 32);
                lrow[r] = lrow[r] * scl + ps;
                mrow[r] = mnew;
                y0[r] *= scl; y1[r] *= scl; y2[r] *= scl; y3[r] *= scl;
                Ps[wid][8 * half + r][l16]      = (bf16_t)p0;
                Ps[wid][8 * half + r][16 + l16] = (bf16_t)p1;
            }
            // order wave-local P stores before fragment reload (split DS counter)
            WAIT_DS0();
            // P as A-fragment (16 rows x 32 keys)
            bf16x8 plo = *(const bf16x8*)&Ps[wid][l16][half * 8];
            bf16x8 phi = *(const bf16x8*)&Ps[wid][l16][16 + half * 8];
            bf16x16 pf = SHUF16(plo, phi);
            // V B-fragments straight from row-major LDS tile (strided 16-bit reads)
#pragma unroll
            for (int n = 0; n < 4; n++) {
                bf16x16 vfv;
#pragma unroll
                for (int e = 0; e < 16; e++) {
                    int k = ((e >> 3) << 4) + half * 8 + (e & 7);
                    vfv[e] = Vs[buf][k][n * 16 + l16];
                }
                if (n == 0) y0 = __builtin_amdgcn_wmma_f32_16x16x32_bf16(false, pf, false, vfv, (short)0, y0, false, false);
                if (n == 1) y1 = __builtin_amdgcn_wmma_f32_16x16x32_bf16(false, pf, false, vfv, (short)0, y1, false, false);
                if (n == 2) y2 = __builtin_amdgcn_wmma_f32_16x16x32_bf16(false, pf, false, vfv, (short)0, y2, false, false);
                if (n == 3) y3 = __builtin_amdgcn_wmma_f32_16x16x32_bf16(false, pf, false, vfv, (short)0, y3, false, false);
            }
        }
        __syncthreads();
    }

    // write y[b,t,h,d]
#pragma unroll
    for (int r = 0; r < 8; r++) {
        int t = qbase + 8 * half + r;
        float inv = 1.0f / lrow[r];
        float* yr = Y + (((size_t)(b_ * Tt + t) * Hh + h) * Dd);
        yr[0 * 16 + l16] = y0[r] * inv;
        yr[1 * 16 + l16] = y1[r] * inv;
        yr[2 * 16 + l16] = y2[r] * inv;
        yr[3 * 16 + l16] = y3[r] * inv;
    }
}

// ---------------- out-fact head mix + residual: xt_out = xt + mix(y) ------
__global__ void k_outmix(const float* __restrict__ xt, const float* __restrict__ of,
                         const float* __restrict__ y, float* __restrict__ xt_out) {
    size_t i = (size_t)blockIdx.x * blockDim.x + threadIdx.x;   // over B*T*C
    int d  = (int)(i & (Dd - 1));
    int hh = (int)((i >> 6) & (Hh - 1));
    int bt = (int)(i >> 10);
    const float* yrow = y + (size_t)bt * Cc;   // [bt][h][d]
    float acc = 0.f;
#pragma unroll
    for (int j = 0; j < Hh; j++) acc += of[hh * Hh + j] * yrow[j * Dd + d];
    xt_out[i] = xt[i] + acc;
}

// ---------------- launcher ----------------
extern "C" void kernel_launch(void* const* d_in, const int* in_sizes, int n_in,
                              void* d_out, int out_size, void* d_ws, size_t ws_size,
                              hipStream_t stream) {
    const float* xt       = (const float*)d_in[0];
    const float* xe       = (const float*)d_in[1];
    const float* qk_w     = (const float*)d_in[2];
    const float* qk_b     = (const float*)d_in[3];
    const float* v_fact   = (const float*)d_in[4];
    const float* out_fact = (const float*)d_in[5];
    const float* ln1_w    = (const float*)d_in[6];
    const float* ln1_b    = (const float*)d_in[7];
    const float* ln2_w    = (const float*)d_in[8];
    const float* ln2_b    = (const float*)d_in[9];
    const float* fc_w     = (const float*)d_in[10];
    const float* fc_b     = (const float*)d_in[11];
    const float* proj_w   = (const float*)d_in[12];
    const float* proj_b   = (const float*)d_in[13];

    float* xt_out = (float*)d_out;
    float* xe_out = (float*)d_out + (size_t)BT * Cc;

    // workspace carve-up
    char* w = (char*)d_ws;
    auto take = [&](size_t bytes) { char* p = w; w += (bytes + 255) & ~(size_t)255; return p; };
    bf16_t* qkw_b   = (bf16_t*)take((size_t)2 * Cc * Cc * 2);        // 4 MB
    bf16_t* fcw_b   = (bf16_t*)take((size_t)DFFf * Cc * 2);          // 8 MB
    bf16_t* projw_b = (bf16_t*)take((size_t)Cc * DFFf * 2);          // 8 MB
    bf16_t* xnorm   = (bf16_t*)take((size_t)BT * Cc * 2);            // 8 MB
    bf16_t* qbh     = (bf16_t*)take((size_t)BT * Cc * 2);            // 8 MB
    bf16_t* kbh     = (bf16_t*)take((size_t)BT * Cc * 2);            // 8 MB
    bf16_t* vbh     = (bf16_t*)take((size_t)BT * Cc * 2);            // 8 MB
    float*  ybuf    = (float*) take((size_t)BT * Cc * 4);            // 16 MB
    bf16_t* xnorm2  = (bf16_t*)take((size_t)BT * Cc * 2);            // 8 MB
    bf16_t* hbuf    = (bf16_t*)take((size_t)BT * DFFf * 2);          // 32 MB

    // 1) weights -> bf16
    k_cvt_bf16<<<(2 * Cc * Cc + 255) / 256, 256, 0, stream>>>(qk_w, qkw_b, 2 * Cc * Cc);
    k_cvt_bf16<<<(DFFf * Cc + 255) / 256, 256, 0, stream>>>(fc_w, fcw_b, DFFf * Cc);
    k_cvt_bf16<<<(Cc * DFFf + 255) / 256, 256, 0, stream>>>(proj_w, projw_b, Cc * DFFf);

    // 2) x_norm = LN(xt + xe)
    k_add_ln<<<BT, 256, 0, stream>>>(xt, xe, ln1_w, ln1_b, xnorm);

    // 3) qk GEMM -> q (scaled 1/sqrt(D)), k scattered to [B,H,T,D]
    k_gemm<<<dim3(BT / 128, (2 * Cc) / 64), 256, 0, stream>>>(
        xnorm, qkw_b, qk_b, BT, 2 * Cc, Cc, /*mode*/0, 0.125f,
        qbh, kbh, nullptr, nullptr, nullptr);

    // 4) v = head-mix of xt
    k_vproj<<<(BT * Cc + 255) / 256, 256, 0, stream>>>(xt, v_fact, vbh);

    // 5) flash attention with ALiBi + causal mask
    k_attn<<<dim3(Tt / 64, Bb * Hh), 128, 0, stream>>>(qbh, kbh, vbh, ybuf);

    // 6) xt_out = xt + out_fact-mix(y)   (first half of d_out)
    k_outmix<<<(BT * Cc + 255) / 256, 256, 0, stream>>>(xt, out_fact, ybuf, xt_out);

    // 7) x_norm2 = LN(xt_out + xe)
    k_add_ln<<<BT, 256, 0, stream>>>(xt_out, xe, ln2_w, ln2_b, xnorm2);

    // 8) h = gelu(x_norm2 @ fc_w.T + fc_b)
    k_gemm<<<dim3(BT / 128, DFFf / 64), 256, 0, stream>>>(
        xnorm2, fcw_b, fc_b, BT, DFFf, Cc, /*mode*/1, 1.0f,
        nullptr, nullptr, hbuf, nullptr, nullptr);

    // 9) xe_out = xe + (h @ proj_w.T + proj_b)   (second half of d_out)
    k_gemm<<<dim3(BT / 128, Cc / 64), 256, 0, stream>>>(
        hbuf, projw_b, proj_b, BT, Cc, DFFf, /*mode*/2, 1.0f,
        nullptr, nullptr, nullptr, xe, xe_out);
}
